// PointNetPlusPlus_9259949491068
// MI455X (gfx1250) — matrix operand order repaired and checked by reading, not compiled, not run on hardware
//
#include <hip/hip_runtime.h>
#include <hip/hip_bf16.h>

// PointNet++ classification forward for MI455X (gfx1250, wave32).
// Heavy per-point MLP layers run on v_wmma_f32_16x16x32_f16 (f16 in, f32 acc).
// Each wave computes a 16x64 output strip (4 N-tiles) so the A fragment is
// loaded once per K-step and reused by 4 WMMAs (4 independent accum chains).
// Max-pools are fused into the last GEMM of each stage via integer atomicMax on
// float bit patterns (valid: post-ReLU values are >= 0, init 0).
// Workspace requirement: ~176 MB.

typedef __attribute__((ext_vector_type(16))) _Float16 v16h;
typedef __attribute__((ext_vector_type(8)))  _Float16 v8h;
typedef __attribute__((ext_vector_type(8)))  float    v8f;

#define BQ   32
#define NPTS 4096

// A frag: lane&15 -> row m0+(lane&15), 16 contiguous K halves at k+kh*16
// B frag: lane&15 -> col n0+(lane&15), 16 contiguous K halves at k+kh*16
__device__ __forceinline__ v16h load_afrag(const _Float16* __restrict__ arow,
                                           int k, int kh) {
  v8h alo = *(const v8h*)(arow + k + kh * 16);
  v8h ahi = *(const v8h*)(arow + k + kh * 16 + 8);
  return __builtin_shufflevector(alo, ahi, 0,1,2,3,4,5,6,7,8,9,10,11,12,13,14,15);
}

// GEMM + optional bias + ReLU, f16 output. block = (32,4): one wave per
// 16x64 strip. N must be a multiple of 64, K a multiple of 32.
__global__ void gemm_relu_f16(const _Float16* __restrict__ A,
                              const _Float16* __restrict__ W,
                              const float* __restrict__ bias,
                              _Float16* __restrict__ O,
                              int M, int N, int K, int lda, int ldw) {
  const int lane = threadIdx.x;
  long strip = (long)blockIdx.x * blockDim.y + threadIdx.y;
  long nstrips = N / 64;
  long tstrips = (long)(M / 16) * nstrips;
  if (strip >= tstrips) return;               // whole wave exits together
  long m0 = (strip / nstrips) * 16;
  long n0 = (strip % nstrips) * 64;
  const int kh = lane >> 4, l15 = lane & 15;
  const _Float16* arow = A + (m0 + l15) * (long)lda;
  const _Float16* bcol = W + (n0 + l15) * (long)ldw;
  v8f acc0 = {}, acc1 = {}, acc2 = {}, acc3 = {};
  for (int k = 0; k < K; k += 32) {
    v16h a  = load_afrag(arow, k, kh);
    v16h b0 = *(const v16h*)(bcol + 0  * (long)ldw + k + kh * 16);
    v16h b1 = *(const v16h*)(bcol + 16 * (long)ldw + k + kh * 16);
    v16h b2 = *(const v16h*)(bcol + 32 * (long)ldw + k + kh * 16);
    v16h b3 = *(const v16h*)(bcol + 48 * (long)ldw + k + kh * 16);
    acc0 = __builtin_amdgcn_wmma_f32_16x16x32_f16(false, a, false, b0, (short)0, acc0, false, false);
    acc1 = __builtin_amdgcn_wmma_f32_16x16x32_f16(false, a, false, b1, (short)0, acc1, false, false);
    acc2 = __builtin_amdgcn_wmma_f32_16x16x32_f16(false, a, false, b2, (short)0, acc2, false, false);
    acc3 = __builtin_amdgcn_wmma_f32_16x16x32_f16(false, a, false, b3, (short)0, acc3, false, false);
  }
  const int col = lane & 15, rbase = (lane >> 4) * 8;
  v8f accs[4] = {acc0, acc1, acc2, acc3};
#pragma unroll
  for (int j = 0; j < 4; ++j) {
    long nc = n0 + j * 16 + col;
    float bv = bias ? bias[nc] : 0.f;
#pragma unroll
    for (int r = 0; r < 8; ++r) {
      float v = accs[j][r] + bv;
      v = v > 0.f ? v : 0.f;
      O[(m0 + rbase + r) * (long)N + nc] = (_Float16)v;
    }
  }
}

// GEMM + bias + ReLU + max-pool over `group` consecutive rows into f32 output.
// group is a multiple of 16 so a strip never straddles pool groups.
__global__ void gemm_relu_maxpool(const _Float16* __restrict__ A,
                                  const _Float16* __restrict__ W,
                                  const float* __restrict__ bias,
                                  float* __restrict__ Opool,
                                  int M, int N, int K, int lda, int ldw, int group) {
  const int lane = threadIdx.x;
  long strip = (long)blockIdx.x * blockDim.y + threadIdx.y;
  long nstrips = N / 64;
  long tstrips = (long)(M / 16) * nstrips;
  if (strip >= tstrips) return;
  long m0 = (strip / nstrips) * 16;
  long n0 = (strip % nstrips) * 64;
  const int kh = lane >> 4, l15 = lane & 15;
  const _Float16* arow = A + (m0 + l15) * (long)lda;
  const _Float16* bcol = W + (n0 + l15) * (long)ldw;
  v8f acc0 = {}, acc1 = {}, acc2 = {}, acc3 = {};
  for (int k = 0; k < K; k += 32) {
    v16h a  = load_afrag(arow, k, kh);
    v16h b0 = *(const v16h*)(bcol + 0  * (long)ldw + k + kh * 16);
    v16h b1 = *(const v16h*)(bcol + 16 * (long)ldw + k + kh * 16);
    v16h b2 = *(const v16h*)(bcol + 32 * (long)ldw + k + kh * 16);
    v16h b3 = *(const v16h*)(bcol + 48 * (long)ldw + k + kh * 16);
    acc0 = __builtin_amdgcn_wmma_f32_16x16x32_f16(false, a, false, b0, (short)0, acc0, false, false);
    acc1 = __builtin_amdgcn_wmma_f32_16x16x32_f16(false, a, false, b1, (short)0, acc1, false, false);
    acc2 = __builtin_amdgcn_wmma_f32_16x16x32_f16(false, a, false, b2, (short)0, acc2, false, false);
    acc3 = __builtin_amdgcn_wmma_f32_16x16x32_f16(false, a, false, b3, (short)0, acc3, false, false);
  }
  const int col = lane & 15;
  long orow = m0 / group;
  v8f accs[4] = {acc0, acc1, acc2, acc3};
#pragma unroll
  for (int j = 0; j < 4; ++j) {
    long nc = n0 + j * 16 + col;
    float bv = bias ? bias[nc] : 0.f;
    float mx = 0.f;
#pragma unroll
    for (int r = 0; r < 8; ++r) {
      float v = accs[j][r] + bv;
      v = v > 0.f ? v : 0.f;
      mx = fmaxf(mx, v);
    }
    atomicMax((int*)Opool + orow * (long)N + nc, __float_as_int(mx));
  }
}

// ---------------------------------------------------------------------------
// Support kernels
// ---------------------------------------------------------------------------
__global__ void zero_kernel(float* p, long n) {
  long i = (long)blockIdx.x * 256 + threadIdx.x;
  if (i < n) p[i] = 0.f;
}

__global__ void convert_f16_kernel(const float* __restrict__ src,
                                   _Float16* __restrict__ dst,
                                   int Nr, int K, int Kpad) {
  long i = (long)blockIdx.x * 256 + threadIdx.x;
  long total = (long)Nr * Kpad;
  if (i >= total) return;
  long r = i / Kpad;
  int  c = (int)(i - r * Kpad);
  dst[i] = (c < K) ? (_Float16)src[r * (long)K + c] : (_Float16)0.f;
}

// center to mean, scale by max L2 norm (per batch)
__global__ void normalize_kernel(const float* __restrict__ xyz, float* __restrict__ out) {
  const int b = blockIdx.x, tid = threadIdx.x;
  const float* p = xyz + (long)b * NPTS * 3;
  float* o = out + (long)b * NPTS * 3;
  __shared__ float red[256];
  __shared__ float meanv[3];
  __shared__ float sinv;
  for (int c = 0; c < 3; ++c) {
    float s = 0.f;
    for (int i = tid; i < NPTS; i += 256) s += p[i * 3 + c];
    red[tid] = s; __syncthreads();
    for (int st = 128; st > 0; st >>= 1) {
      if (tid < st) red[tid] += red[tid + st];
      __syncthreads();
    }
    if (tid == 0) meanv[c] = red[0] / (float)NPTS;
    __syncthreads();
  }
  float mx = 0.f;
  for (int i = tid; i < NPTS; i += 256) {
    float dx = p[i*3+0]-meanv[0], dy = p[i*3+1]-meanv[1], dz = p[i*3+2]-meanv[2];
    mx = fmaxf(mx, dx*dx + dy*dy + dz*dz);
  }
  red[tid] = mx; __syncthreads();
  for (int st = 128; st > 0; st >>= 1) {
    if (tid < st) red[tid] = fmaxf(red[tid], red[tid + st]);
    __syncthreads();
  }
  if (tid == 0) sinv = 1.f / sqrtf(red[0]);
  __syncthreads();
  for (int i = tid; i < NPTS; i += 256) {
    o[i*3+0] = (p[i*3+0]-meanv[0]) * sinv;
    o[i*3+1] = (p[i*3+1]-meanv[1]) * sinv;
    o[i*3+2] = (p[i*3+2]-meanv[2]) * sinv;
  }
}

// per-point 3->64 MLP (+bias) with ReLU, f16 out. one thread per (row, o<64)
__global__ void mlp3_kernel(const float* __restrict__ pts, const float* __restrict__ w,
                            const float* __restrict__ bias, _Float16* __restrict__ out,
                            long rows) {
  long idx = (long)blockIdx.x * 256 + threadIdx.x;
  if (idx >= rows * 64) return;
  long row = idx >> 6; int o = (int)(idx & 63);
  float x = pts[row*3+0], y = pts[row*3+1], z = pts[row*3+2];
  float v = x*w[o*3+0] + y*w[o*3+1] + z*w[o*3+2] + (bias ? bias[o] : 0.f);
  out[idx] = (_Float16)(v > 0.f ? v : 0.f);
}

// farthest point sampling, one block per batch; distance array in LDS
__global__ void fps_kernel(const float* __restrict__ xyz, int Np, int npoint,
                           int* __restrict__ fps_idx, float* __restrict__ new_xyz) {
  const int b = blockIdx.x, tid = threadIdx.x;
  const float* p = xyz + (long)b * Np * 3;
  __shared__ float dist[4096];
  __shared__ float rd[256];
  __shared__ int   ri[256];
  __shared__ int   s_far;
  for (int i = tid; i < Np; i += 256) dist[i] = 1e10f;
  if (tid == 0) s_far = 0;
  __syncthreads();
  for (int it = 0; it < npoint; ++it) {
    int far = s_far;
    float cx = p[far*3+0], cy = p[far*3+1], cz = p[far*3+2];
    if (tid == 0) {
      fps_idx[b * npoint + it] = far;
      new_xyz[((long)b * npoint + it)*3 + 0] = cx;
      new_xyz[((long)b * npoint + it)*3 + 1] = cy;
      new_xyz[((long)b * npoint + it)*3 + 2] = cz;
    }
    float bd = -1.f; int bi = 0;
    for (int i = tid; i < Np; i += 256) {
      float dx = p[i*3+0]-cx, dy = p[i*3+1]-cy, dz = p[i*3+2]-cz;
      float d = dx*dx + dy*dy + dz*dz;
      float dd = dist[i];
      d = d < dd ? d : dd;
      dist[i] = d;
      if (d > bd) { bd = d; bi = i; }
    }
    rd[tid] = bd; ri[tid] = bi;
    __syncthreads();
    for (int st = 128; st > 0; st >>= 1) {
      if (tid < st) {
        if (rd[tid+st] > rd[tid] || (rd[tid+st] == rd[tid] && ri[tid+st] < ri[tid])) {
          rd[tid] = rd[tid+st]; ri[tid] = ri[tid+st];
        }
      }
      __syncthreads();
    }
    if (tid == 0) s_far = ri[0];
    __syncthreads();
  }
}

// ball query: one wave32 per centroid; in-order compaction via ballot
__global__ void ballquery_kernel(const float* __restrict__ xyz,
                                 const float* __restrict__ cen,
                                 int Np, int S, int nsample, float r2,
                                 int* __restrict__ outidx) {
  int gw = (blockIdx.x * blockDim.x + threadIdx.x) >> 5;
  int lane = threadIdx.x & 31;
  if (gw >= BQ * S) return;
  int b = gw / S;
  const float* p = xyz + (long)b * Np * 3;
  float cx = cen[(long)gw*3+0], cy = cen[(long)gw*3+1], cz = cen[(long)gw*3+2];
  int* out = outidx + (long)gw * nsample;
  int count = 0, first = -1;
  for (int base = 0; base < Np && count < nsample; base += 32) {
    int i = base + lane;
    float dx = p[i*3+0]-cx, dy = p[i*3+1]-cy, dz = p[i*3+2]-cz;
    bool pred = (dx*dx + dy*dy + dz*dz) <= r2;
    unsigned mask = (unsigned)__ballot(pred);
    if (first < 0 && mask != 0u) first = base + __builtin_ctz(mask);
    unsigned prior = mask & ((1u << lane) - 1u);
    int pos = count + __popc(prior);
    if (pred && pos < nsample) out[pos] = i;
    count += __popc(mask);
  }
  if (count < nsample) {
    int fill = (first < 0) ? 0 : first;
    for (int q = count + lane; q < nsample; q += 32) out[q] = fill;
  }
}

__global__ void apply_transform_kernel(const float* __restrict__ xyz,
                                       const float* __restrict__ tr9,
                                       float* __restrict__ out) {
  long i = (long)blockIdx.x * 256 + threadIdx.x;
  if (i >= (long)BQ * NPTS) return;
  int b = (int)(i / NPTS);
  const float* t = tr9 + (long)b * 9;
  float x = xyz[i*3+0], y = xyz[i*3+1], z = xyz[i*3+2];
  out[i*3+0] = (t[0]+1.f)*x + t[1]*y + t[2]*z;
  out[i*3+1] = t[3]*x + (t[4]+1.f)*y + t[5]*z;
  out[i*3+2] = t[6]*x + t[7]*y + (t[8]+1.f)*z;
}

// SA1 layer1: gather+center (K=3) MLP to 64, f16 out. rows = B*512*32
__global__ void sa1_l1_kernel(const float* __restrict__ xyz,
                              const float* __restrict__ cen,
                              const int* __restrict__ bidx,
                              const float* __restrict__ w,
                              _Float16* __restrict__ out) {
  long idx = (long)blockIdx.x * 256 + threadIdx.x;
  const long rows = (long)BQ * 512 * 32;
  if (idx >= rows * 64) return;
  long row = idx >> 6; int o = (int)(idx & 63);
  int b = (int)(row >> 14);          // / (512*32)
  long srow = row >> 5;              // b*512 + s
  long pt = bidx[row];
  float dx = xyz[((long)b*NPTS + pt)*3+0] - cen[srow*3+0];
  float dy = xyz[((long)b*NPTS + pt)*3+1] - cen[srow*3+1];
  float dz = xyz[((long)b*NPTS + pt)*3+2] - cen[srow*3+2];
  float v = dx*w[o*3+0] + dy*w[o*3+1] + dz*w[o*3+2];
  out[idx] = (_Float16)(v > 0.f ? v : 0.f);
}

// SA2 gather: [xyz diff(3) | l1 feats(128) | zero pad(29)] -> f16 (rows x 160)
__global__ void sa2_gather_kernel(const float* __restrict__ l1xyz,
                                  const float* __restrict__ cen2,
                                  const int* __restrict__ bidx,
                                  const float* __restrict__ l1f,
                                  _Float16* __restrict__ out) {
  long idx = (long)blockIdx.x * 256 + threadIdx.x;
  const long rows = (long)BQ * 128 * 64;
  if (idx >= rows * 160) return;
  long row = idx / 160;
  int c = (int)(idx - row * 160);
  int b = (int)(row >> 13);          // / (128*64)
  long srow = row >> 6;              // b*128 + s
  int pt = bidx[row];
  float v;
  if (c < 3)        v = l1xyz[((long)b*512 + pt)*3 + c] - cen2[srow*3 + c];
  else if (c < 131) v = l1f[((long)b*512 + pt)*128 + (c - 3)];
  else              v = 0.f;
  out[idx] = (_Float16)v;
}

// small dense FC (scalar), optional bias / relu
__global__ void fc_kernel(const float* __restrict__ A, const float* __restrict__ W,
                          const float* __restrict__ bias, float* __restrict__ O,
                          int M, int N, int K, int do_relu) {
  int idx = blockIdx.x * 256 + threadIdx.x;
  if (idx >= M * N) return;
  int m = idx / N, n = idx % N;
  const float* a = A + (long)m * K;
  const float* w = W + (long)n * K;
  float acc = bias ? bias[n] : 0.f;
  for (int k = 0; k < K; ++k) acc += a[k] * w[k];
  if (do_relu) acc = acc > 0.f ? acc : 0.f;
  O[idx] = acc;
}

__global__ void log_softmax_kernel(const float* __restrict__ logits, float* __restrict__ out) {
  int b = threadIdx.x;
  if (b >= BQ) return;
  const float* l = logits + b * 10;
  float m = l[0];
  for (int c = 1; c < 10; ++c) m = fmaxf(m, l[c]);
  float s = 0.f;
  for (int c = 0; c < 10; ++c) s += expf(l[c] - m);
  float lse = m + logf(s);
  for (int c = 0; c < 10; ++c) out[b * 10 + c] = l[c] - lse;
}

// ---------------------------------------------------------------------------
extern "C" void kernel_launch(void* const* d_in, const int* in_sizes, int n_in,
                              void* d_out, int out_size, void* d_ws, size_t ws_size,
                              hipStream_t stream) {
  const float* xyz     = (const float*)d_in[0];
  const float* tw1     = (const float*)d_in[1];
  const float* tb1     = (const float*)d_in[2];
  const float* tw2     = (const float*)d_in[3];
  const float* tb2     = (const float*)d_in[4];
  const float* tw3     = (const float*)d_in[5];
  const float* tb3     = (const float*)d_in[6];
  const float* tfc1w   = (const float*)d_in[7];
  const float* tfc1b   = (const float*)d_in[8];
  const float* tfc2w   = (const float*)d_in[9];
  const float* tfc2b   = (const float*)d_in[10];
  const float* tfc3w   = (const float*)d_in[11];
  const float* tfc3b   = (const float*)d_in[12];
  const float* s1w1    = (const float*)d_in[13];
  const float* s1w2    = (const float*)d_in[14];
  const float* s1w3    = (const float*)d_in[15];
  const float* s2w1    = (const float*)d_in[16];
  const float* s2w2    = (const float*)d_in[17];
  const float* s2w3    = (const float*)d_in[18];
  const float* gw1     = (const float*)d_in[19];
  const float* gw2     = (const float*)d_in[20];
  const float* fc1w    = (const float*)d_in[21];
  const float* fc2w    = (const float*)d_in[22];
  const float* fc3w    = (const float*)d_in[23];
  const float* fc3b    = (const float*)d_in[24];

  char* base = (char*)d_ws;
  size_t off = 0;
  auto alloc = [&](size_t bytes) -> void* {
    off = (off + 255) & ~(size_t)255;
    void* p = base + off;
    off += bytes;
    return p;
  };

  float* xyz_n  = (float*)alloc((size_t)BQ*NPTS*3*4);
  float* xyz_t  = (float*)alloc((size_t)BQ*NPTS*3*4);
  float* h3max  = (float*)alloc((size_t)BQ*1024*4);
  float* t1     = (float*)alloc((size_t)BQ*512*4);
  float* t2     = (float*)alloc((size_t)BQ*256*4);
  float* trans  = (float*)alloc((size_t)BQ*9*4);
  int*   fps1   = (int*)  alloc((size_t)BQ*512*4);
  float* nxyz1  = (float*)alloc((size_t)BQ*512*3*4);
  int*   bidx1  = (int*)  alloc((size_t)BQ*512*32*4);
  float* l1f    = (float*)alloc((size_t)BQ*512*128*4);
  int*   fps2   = (int*)  alloc((size_t)BQ*128*4);
  float* nxyz2  = (float*)alloc((size_t)BQ*128*3*4);
  int*   bidx2  = (int*)  alloc((size_t)BQ*128*64*4);
  float* l2f    = (float*)alloc((size_t)BQ*128*256*4);
  float* gmax   = (float*)alloc((size_t)BQ*1024*4);
  float* f1     = (float*)alloc((size_t)BQ*512*4);
  float* f2     = (float*)alloc((size_t)BQ*256*4);
  float* logits = (float*)alloc((size_t)BQ*10*4);
  _Float16* wt2h  = (_Float16*)alloc((size_t)128*64*2);
  _Float16* wt3h  = (_Float16*)alloc((size_t)1024*128*2);
  _Float16* sw2h  = (_Float16*)alloc((size_t)64*64*2);
  _Float16* sw3h  = (_Float16*)alloc((size_t)128*64*2);
  _Float16* s2w1h = (_Float16*)alloc((size_t)128*160*2);
  _Float16* s2w2h = (_Float16*)alloc((size_t)128*128*2);
  _Float16* s2w3h = (_Float16*)alloc((size_t)256*128*2);
  _Float16* gw1h  = (_Float16*)alloc((size_t)512*256*2);
  _Float16* gw2h  = (_Float16*)alloc((size_t)1024*512*2);
  _Float16* bufA  = (_Float16*)alloc((size_t)262144*160*2);  // 80 MB ping
  _Float16* bufB  = (_Float16*)alloc((size_t)524288*64*2);   // 64 MB pong

  dim3 wblk(32, 4);
  auto gemm_grid = [](long M, long N) { return (unsigned)(((M/16)*(N/64) + 3) / 4); };
  auto flat_grid = [](long n) { return (unsigned)((n + 255) / 256); };

  // weight conversions to f16 (SA2 layer1 zero-padded K 131->160)
  convert_f16_kernel<<<flat_grid(128*64),   256, 0, stream>>>(tw2,  wt2h, 128, 64, 64);
  convert_f16_kernel<<<flat_grid(1024*128), 256, 0, stream>>>(tw3,  wt3h, 1024, 128, 128);
  convert_f16_kernel<<<flat_grid(64*64),    256, 0, stream>>>(s1w2, sw2h, 64, 64, 64);
  convert_f16_kernel<<<flat_grid(128*64),   256, 0, stream>>>(s1w3, sw3h, 128, 64, 64);
  convert_f16_kernel<<<flat_grid(128*160),  256, 0, stream>>>(s2w1, s2w1h, 128, 131, 160);
  convert_f16_kernel<<<flat_grid(128*128),  256, 0, stream>>>(s2w2, s2w2h, 128, 128, 128);
  convert_f16_kernel<<<flat_grid(256*128),  256, 0, stream>>>(s2w3, s2w3h, 256, 128, 128);
  convert_f16_kernel<<<flat_grid(512*256),  256, 0, stream>>>(gw1,  gw1h, 512, 256, 256);
  convert_f16_kernel<<<flat_grid(1024*512), 256, 0, stream>>>(gw2,  gw2h, 1024, 512, 512);

  // normalize
  normalize_kernel<<<BQ, 256, 0, stream>>>(xyz, xyz_n);

  // ---- T-Net ----
  const long TP = (long)BQ * NPTS;                 // 131072 points
  mlp3_kernel<<<flat_grid(TP*64), 256, 0, stream>>>(xyz_n, tw1, tb1, bufB, TP);
  gemm_relu_f16<<<gemm_grid(TP, 128), wblk, 0, stream>>>(bufB, wt2h, tb2, bufA,
                                                         (int)TP, 128, 64, 64, 64);
  zero_kernel<<<flat_grid(BQ*1024), 256, 0, stream>>>(h3max, BQ*1024);
  gemm_relu_maxpool<<<gemm_grid(TP, 1024), wblk, 0, stream>>>(bufA, wt3h, tb3, h3max,
                                                              (int)TP, 1024, 128, 128, 128, NPTS);
  fc_kernel<<<flat_grid(BQ*512), 256, 0, stream>>>(h3max, tfc1w, tfc1b, t1, BQ, 512, 1024, 1);
  fc_kernel<<<flat_grid(BQ*256), 256, 0, stream>>>(t1, tfc2w, tfc2b, t2, BQ, 256, 512, 1);
  fc_kernel<<<flat_grid(BQ*9),   256, 0, stream>>>(t2, tfc3w, tfc3b, trans, BQ, 9, 256, 0);
  apply_transform_kernel<<<flat_grid(TP), 256, 0, stream>>>(xyz_n, trans, xyz_t);

  // ---- SA1: 4096 -> 512 centroids, nsample 32, r=0.2 ----
  fps_kernel<<<BQ, 256, 0, stream>>>(xyz_t, NPTS, 512, fps1, nxyz1);
  ballquery_kernel<<<(BQ*512)/4, 128, 0, stream>>>(xyz_t, nxyz1, NPTS, 512, 32, 0.04f, bidx1);
  const long R1 = (long)BQ * 512 * 32;             // 524288 grouped rows
  sa1_l1_kernel<<<flat_grid(R1*64), 256, 0, stream>>>(xyz_t, nxyz1, bidx1, s1w1, bufB);
  gemm_relu_f16<<<gemm_grid(R1, 64), wblk, 0, stream>>>(bufB, sw2h, nullptr, bufA,
                                                        (int)R1, 64, 64, 64, 64);
  zero_kernel<<<flat_grid((long)BQ*512*128), 256, 0, stream>>>(l1f, (long)BQ*512*128);
  gemm_relu_maxpool<<<gemm_grid(R1, 128), wblk, 0, stream>>>(bufA, sw3h, nullptr, l1f,
                                                             (int)R1, 128, 64, 64, 64, 32);

  // ---- SA2: 512 -> 128 centroids, nsample 64, r=0.4, feats 131 (pad 160) ----
  fps_kernel<<<BQ, 256, 0, stream>>>(nxyz1, 512, 128, fps2, nxyz2);
  ballquery_kernel<<<(BQ*128)/4, 128, 0, stream>>>(nxyz1, nxyz2, 512, 128, 64, 0.16f, bidx2);
  const long R2 = (long)BQ * 128 * 64;             // 262144 grouped rows
  sa2_gather_kernel<<<flat_grid(R2*160), 256, 0, stream>>>(nxyz1, nxyz2, bidx2, l1f, bufA);
  gemm_relu_f16<<<gemm_grid(R2, 128), wblk, 0, stream>>>(bufA, s2w1h, nullptr, bufB,
                                                         (int)R2, 128, 160, 160, 160);
  gemm_relu_f16<<<gemm_grid(R2, 128), wblk, 0, stream>>>(bufB, s2w2h, nullptr, bufA,
                                                         (int)R2, 128, 128, 128, 128);
  zero_kernel<<<flat_grid((long)BQ*128*256), 256, 0, stream>>>(l2f, (long)BQ*128*256);
  gemm_relu_maxpool<<<gemm_grid(R2, 256), wblk, 0, stream>>>(bufA, s2w3h, nullptr, l2f,
                                                             (int)R2, 256, 128, 128, 128, 64);

  // ---- Global MLP 256->512->1024, max over 128 points ----
  const long RG = (long)BQ * 128;                  // 4096 rows
  convert_f16_kernel<<<flat_grid(RG*256), 256, 0, stream>>>(l2f, bufB, (int)RG, 256, 256);
  gemm_relu_f16<<<gemm_grid(RG, 512), wblk, 0, stream>>>(bufB, gw1h, nullptr, bufA,
                                                         (int)RG, 512, 256, 256, 256);
  zero_kernel<<<flat_grid(BQ*1024), 256, 0, stream>>>(gmax, BQ*1024);
  gemm_relu_maxpool<<<gemm_grid(RG, 1024), wblk, 0, stream>>>(bufA, gw2h, nullptr, gmax,
                                                              (int)RG, 1024, 512, 512, 512, 128);

  // ---- Head ----
  fc_kernel<<<flat_grid(BQ*512), 256, 0, stream>>>(gmax, fc1w, nullptr, f1, BQ, 512, 1024, 1);
  fc_kernel<<<flat_grid(BQ*256), 256, 0, stream>>>(f1, fc2w, nullptr, f2, BQ, 256, 512, 1);
  fc_kernel<<<flat_grid(BQ*10),  256, 0, stream>>>(f2, fc3w, fc3b, logits, BQ, 10, 256, 0);
  log_softmax_kernel<<<1, 32, 0, stream>>>(logits, (float*)d_out);

  (void)in_sizes; (void)n_in; (void)out_size; (void)ws_size;
}
